// FrameBlock_48326972015014
// MI455X (gfx1250) — matrix operand order repaired and checked by reading, not compiled, not run on hardware
//
#include <hip/hip_runtime.h>

// ---------------------------------------------------------------------------
// FrameBlock (graph-diffusion GRU), MI455X / gfx1250, wave32 + WMMA bf16.
// BS=8, IN_CH=1, HID=64, V=1024, STEPS=64.
//
// Dominant math per step:
//   h_diff[m=(b,hid), w] = sum_v h[m,v] * graph[w,v]      (512x1024x1024 GEMM)
//   gh[g, (b,w)]         = sum_k W_hh[g,k] * h_diff[b,k,w] (192x64x8192 GEMM)
// Both mapped to v_wmma_f32_16x16x32_bf16, fp32 accumulate.
// hdiff uses 2x2 register blocking (32x32 per wave) -> 16 FLOP/byte of
// cache traffic instead of 8, so the WMMA pipe (not L0/L2) is the limiter.
// ---------------------------------------------------------------------------

typedef __attribute__((ext_vector_type(16))) __bf16 v16bf;
typedef __attribute__((ext_vector_type(8)))  float  v8f;

#define BS     8
#define HID    64
#define NN     1024
#define STEPS  64
#define MROWS  (BS * HID)   // 512

// ---------------------------------------------------------------------------
// Prep: convert graph & W_hh to bf16, broadcast h0 into the fp32/bf16 state.
// ---------------------------------------------------------------------------
__global__ __launch_bounds__(256) void prep_kernel(
    const float* __restrict__ graph, const float* __restrict__ whh,
    const float* __restrict__ h0,
    float* __restrict__ h_f32, __bf16* __restrict__ h_bf,
    __bf16* __restrict__ g_bf, __bf16* __restrict__ whh_bf)
{
    int i = blockIdx.x * blockDim.x + threadIdx.x;
    int stride = gridDim.x * blockDim.x;
    for (int k = i; k < NN * NN; k += stride)
        g_bf[k] = (__bf16)graph[k];
    for (int k = i; k < 3 * HID * HID; k += stride)
        whh_bf[k] = (__bf16)whh[k];
    for (int k = i; k < MROWS * NN; k += stride) {
        float v = h0[k & (HID * NN - 1)];   // broadcast over batch
        h_f32[k] = v;
        h_bf[k]  = (__bf16)v;
    }
}

// ---------------------------------------------------------------------------
// x_hat kernel: x_hat1 = W_init . h + b_init  (preds output), then mask-mix.
// One thread per (b,v); loads coalesced across v.
// ---------------------------------------------------------------------------
__global__ __launch_bounds__(256) void xhat_kernel(
    int t,
    const float* __restrict__ x, const unsigned char* __restrict__ mask,
    const float* __restrict__ Winit, const float* __restrict__ binit,
    const float* __restrict__ h_f32,
    float* __restrict__ xhat_ws, float* __restrict__ preds_out)
{
    int tid = blockIdx.x * blockDim.x + threadIdx.x;   // 0..8191 = b*1024+v
    int b = tid >> 10, v = tid & (NN - 1);
    float s = binit[0];
    const float* hp = h_f32 + (b * HID) * NN + v;
#pragma unroll 8
    for (int h = 0; h < HID; ++h)
        s += Winit[h] * hp[h * NN];
    int o = tid * STEPS + t;                           // (b,0,v,t)
    preds_out[o] = s;
    float xv = x[o];
    xhat_ws[tid] = mask[o] ? xv : s;
}

// ---------------------------------------------------------------------------
// h_diff WMMA kernel: D = H(512x1024) x graph^T, bf16 in / fp32 acc.
// One wave = one 32x32 output block (2x2 tiles of 16x16), K-loop of 32 over v.
// Stores transposed (b, w, hid) fp32 + bf16 so the GRU's B-fragments are
// contiguous over hid.
// ---------------------------------------------------------------------------
__device__ __forceinline__ void store_hd_tile(
    const v8f& acc, int mtile_base, int w,
    float* __restrict__ hdT, __bf16* __restrict__ hdT_bf)
{
    // D layout: lane holds col n (folded into w), rows j + half*8 (+ base).
    int b    = mtile_base >> 6;          // /HID
    int hid0 = mtile_base & (HID - 1);
    float*  pd  = hdT    + (b * NN + w) * HID + hid0;
    __bf16* pdb = hdT_bf + (b * NN + w) * HID + hid0;
    *(v8f*)pd = acc;                     // 32B contiguous over hid
#pragma unroll
    for (int j = 0; j < 8; ++j) pdb[j] = (__bf16)acc[j];
}

__global__ __launch_bounds__(256) void hdiff_kernel(
    const __bf16* __restrict__ h_bf, const __bf16* __restrict__ g_bf,
    float* __restrict__ hdT, __bf16* __restrict__ hdT_bf)
{
    int lane = threadIdx.x & 31;
    int wave = threadIdx.x >> 5;
    int wid  = blockIdx.x * 8 + wave;   // 512 wave-units (32x32 blocks)
    int mT = wid & 15;                  // 16 blocks of 32 over M=512
    int nT = wid >> 4;                  // 32 blocks of 32 over N=1024
    int half = (lane >> 4) & 1;
    int l16  = lane & 15;

    // A: lane holds row m=l16, 16 consecutive K at base=half*16
    const __bf16* pa0 = h_bf + (mT * 32 + l16) * NN + half * 16;
    const __bf16* pa1 = pa0 + 16 * NN;
    // B: lane holds col n=l16, 16 consecutive K at base=half*16 (graph row n)
    const __bf16* pb0 = g_bf + (nT * 32 + l16) * NN + half * 16;
    const __bf16* pb1 = pb0 + 16 * NN;

    v8f c00 = {}, c01 = {}, c10 = {}, c11 = {};
    for (int k0 = 0; k0 < NN; k0 += 32) {
        v16bf a0, a1;
#pragma unroll
        for (int i = 0; i < 8; ++i) {
            a0[i] = pa0[k0 + i]; a0[i + 8] = pa0[k0 + 8 + i];
            a1[i] = pa1[k0 + i]; a1[i + 8] = pa1[k0 + 8 + i];
        }
        v16bf b0 = *(const v16bf*)(pb0 + k0);
        v16bf b1 = *(const v16bf*)(pb1 + k0);
        c00 = __builtin_amdgcn_wmma_f32_16x16x32_bf16(
                  false, a0, false, b0, (short)0, c00, false, false);
        c10 = __builtin_amdgcn_wmma_f32_16x16x32_bf16(
                  false, a1, false, b0, (short)0, c10, false, false);
        c01 = __builtin_amdgcn_wmma_f32_16x16x32_bf16(
                  false, a0, false, b1, (short)0, c01, false, false);
        c11 = __builtin_amdgcn_wmma_f32_16x16x32_bf16(
                  false, a1, false, b1, (short)0, c11, false, false);
    }

    int mb0 = mT * 32 + half * 8;       // tile-row base for c0x
    int mb1 = mb0 + 16;                 // tile-row base for c1x
    int w0  = nT * 32 + l16;            // col for cx0
    int w1  = w0 + 16;                  // col for cx1
    store_hd_tile(c00, mb0, w0, hdT, hdT_bf);
    store_hd_tile(c01, mb0, w1, hdT, hdT_bf);
    store_hd_tile(c10, mb1, w0, hdT, hdT_bf);
    store_hd_tile(c11, mb1, w1, hdT, hdT_bf);
}

// ---------------------------------------------------------------------------
// GRU kernel: per block = 16 nodes x all 8 batches (wave = batch).
// Phase 1: x_g = x_hat . ind_graph^T (VALU dots into LDS).
// Phase 2: gh = W_hh x h_diff via 24 WMMAs/wave (12 M-tiles x 2 K-steps),
//          then lane-local gates, h update, W_out projection.
// ---------------------------------------------------------------------------
__global__ __launch_bounds__(256) void gru_kernel(
    int t,
    const float* __restrict__ xhat_ws, const float* __restrict__ indg,
    const __bf16* __restrict__ whh_bf,
    const __bf16* __restrict__ hdT_bf, const float* __restrict__ hdT,
    const float* __restrict__ Wih, const float* __restrict__ bih,
    const float* __restrict__ bhh,
    const float* __restrict__ Wout, const float* __restrict__ bout,
    float* __restrict__ h_f32, __bf16* __restrict__ h_bf,
    float* __restrict__ imps, float* __restrict__ reprs)
{
    __shared__ float part[256];
    __shared__ float smxg[128];   // [b*16 + wl]
    int w0 = blockIdx.x * 16;
    int tidx = threadIdx.x;

    // ---- Phase 1: x_g[b, w0+wl] (2 threads per pair, halves of v) ----
    {
        int pair = tidx >> 1, vh = tidx & 1;
        int pb = pair >> 4, wl = pair & 15;
        const float* xr = xhat_ws + pb * NN + vh * 512;
        const float* gr = indg + (w0 + wl) * NN + vh * 512;
        float s = 0.f;
        for (int v = 0; v < 512; ++v) s += xr[v] * gr[v];
        part[tidx] = s;
    }
    __syncthreads();
    if (tidx < 128) smxg[tidx] = part[2 * tidx] + part[2 * tidx + 1];
    __syncthreads();

    // ---- Phase 2: WMMA gh + gates (EXEC all ones here) ----
    int lane = tidx & 31;
    int b    = tidx >> 5;            // wave id = batch
    int half = (lane >> 4) & 1;
    int n    = lane & 15;
    int w    = w0 + n;

    // B fragments: column (b,w), contiguous over hid (K), 2 K-steps of 32
    const __bf16* pb = hdT_bf + (b * NN + w) * HID + half * 16;
    v16bf bf0 = *(const v16bf*)(pb);
    v16bf bf1 = *(const v16bf*)(pb + 32);

    v8f acc[12];
#pragma unroll
    for (int mt = 0; mt < 12; ++mt) {
        const __bf16* pa = whh_bf + (mt * 16 + n) * HID + half * 16;
        v16bf a0, a1;
#pragma unroll
        for (int i = 0; i < 8; ++i) {
            a0[i] = pa[i];      a0[i + 8] = pa[i + 8];
            a1[i] = pa[32 + i]; a1[i + 8] = pa[32 + i + 8];
        }
        v8f c = {};
        c = __builtin_amdgcn_wmma_f32_16x16x32_bf16(
                false, a0, false, bf0, (short)0, c, false, false);
        c = __builtin_amdgcn_wmma_f32_16x16x32_bf16(
                false, a1, false, bf1, (short)0, c, false, false);
        acc[mt] = c;
    }

    // Gates: for hid = ht*16 + half*8 + j, rows {hid, 64+hid, 128+hid}
    // live in tiles {ht, 4+ht, 8+ht}, same lane & element j.
    float xg = smxg[b * 16 + n];
    float outp = 0.f;
    const float* hdcol = hdT + (b * NN + w) * HID;
#pragma unroll
    for (int ht = 0; ht < 4; ++ht) {
#pragma unroll
        for (int j = 0; j < 8; ++j) {
            int hid = ht * 16 + half * 8 + j;
            float hr = acc[ht][j]     + bhh[hid];
            float hz = acc[4 + ht][j] + bhh[64 + hid];
            float hn = acc[8 + ht][j] + bhh[128 + hid];
            float ir  = xg * Wih[hid]       + bih[hid];
            float iz  = xg * Wih[64 + hid]  + bih[64 + hid];
            float inn = xg * Wih[128 + hid] + bih[128 + hid];
            float r  = 1.f / (1.f + __expf(-(ir + hr)));
            float z  = 1.f / (1.f + __expf(-(iz + hz)));
            float nn = tanhf(inn + r * hn);
            float hd = hdcol[hid];
            float hnew = (1.f - z) * nn + z * hd;
            int hidx = (b * HID + hid) * NN + w;   // (b,hid,v) state layout
            h_f32[hidx] = hnew;
            h_bf[hidx]  = (__bf16)hnew;
            reprs[hidx * STEPS + t] = hnew;        // (b,hid,v,step)
            outp += Wout[hid] * hnew;
        }
    }
    // reduce W_out projection across the two lane-halves of this column
    outp += __shfl_xor(outp, 16, 32);
    if (half == 0)
        imps[(b * NN + w) * STEPS + t] = outp + bout[0];
}

// ---------------------------------------------------------------------------
// Host side
// ---------------------------------------------------------------------------
static inline size_t alignup(size_t x) { return (x + 255) & ~(size_t)255; }

extern "C" void kernel_launch(void* const* d_in, const int* in_sizes, int n_in,
                              void* d_out, int out_size, void* d_ws, size_t ws_size,
                              hipStream_t stream) {
    (void)in_sizes; (void)n_in; (void)out_size; (void)ws_size;
    const float*         x     = (const float*)d_in[0];
    const unsigned char* mask  = (const unsigned char*)d_in[1];  // bool
    const float*         graph = (const float*)d_in[2];
    const float*         indg  = (const float*)d_in[3];
    const float*         h0    = (const float*)d_in[4];
    const float*         Winit = (const float*)d_in[5];
    const float*         binit = (const float*)d_in[6];
    const float*         Wout  = (const float*)d_in[7];
    const float*         bout  = (const float*)d_in[8];
    const float*         Wih   = (const float*)d_in[9];
    const float*         Whh   = (const float*)d_in[10];
    const float*         bih   = (const float*)d_in[11];
    const float*         bhh   = (const float*)d_in[12];

    // Outputs: imps | preds | reprs, flat fp32
    float* imps  = (float*)d_out;
    float* preds = imps + (size_t)BS * NN * STEPS;           // +524288
    float* reprs = preds + (size_t)BS * NN * STEPS;          // +524288

    // Workspace carve-up
    char* p = (char*)d_ws;
    float*  h_f32  = (float*)p;  p += alignup(sizeof(float)  * MROWS * NN);
    __bf16* h_bf   = (__bf16*)p; p += alignup(sizeof(__bf16) * MROWS * NN);
    __bf16* g_bf   = (__bf16*)p; p += alignup(sizeof(__bf16) * NN * NN);
    __bf16* whh_bf = (__bf16*)p; p += alignup(sizeof(__bf16) * 3 * HID * HID);
    float*  hdT    = (float*)p;  p += alignup(sizeof(float)  * BS * NN * HID);
    __bf16* hdT_bf = (__bf16*)p; p += alignup(sizeof(__bf16) * BS * NN * HID);
    float*  xh_ws  = (float*)p;  p += alignup(sizeof(float)  * BS * NN);

    prep_kernel<<<512, 256, 0, stream>>>(graph, Whh, h0, h_f32, h_bf, g_bf, whh_bf);

    for (int t = 0; t < STEPS; ++t) {
        xhat_kernel<<<(BS * NN) / 256, 256, 0, stream>>>(
            t, x, mask, Winit, binit, h_f32, xh_ws, preds);
        // 512 wave-units of 32x32, 8 waves per block -> 64 blocks
        hdiff_kernel<<<64, 256, 0, stream>>>(h_bf, g_bf, hdT, hdT_bf);
        gru_kernel<<<NN / 16, 256, 0, stream>>>(
            t, xh_ws, indg, whh_bf, hdT_bf, hdT,
            Wih, bih, bhh, Wout, bout,
            h_f32, h_bf, imps, reprs);
    }
}
